// EmbeddingDistance_33784212750477
// MI455X (gfx1250) — matrix operand order repaired and checked by reading, not compiled, not run on hardware
//
#include <hip/hip_runtime.h>

// EmbeddingDistance on MI455X (gfx1250, wave32):
//   out[b] = 1 - dot(mean_s normalize(x1[b,s,:]), mean_s normalize(x2[b,s,:]))
// Shapes fixed by reference: B=16, S=4096, D=256, fp32 in/out.
// Memory-bound (128 MiB streamed once, ~5.5us at 23.3 TB/s). Stage 1 streams
// via async global->LDS double buffering (ASYNCcnt, inline asm
// global_load_async_to_lds_b128); stage 2 computes the 16x16 Gram diagonal
// with v_wmma_f32_16x16x32_f16.

#define ED_B 16
#define ED_S 4096
#define ED_D 256
#define ED_EPS 1e-8f

typedef __attribute__((ext_vector_type(16))) _Float16 v16h;
typedef __attribute__((ext_vector_type(8)))  float    v8f;

#if defined(__HIP_DEVICE_COMPILE__) && defined(__gfx1250__) && \
    __has_builtin(__builtin_amdgcn_s_wait_asynccnt)
#define ED_USE_ASYNC 1
#else
#define ED_USE_ASYNC 0
#endif

#if ED_USE_ASYNC
__device__ __forceinline__ void ed_async16(const float* g, float* l) {
  // 32-bit LDS byte address (AS3 pointers are 32-bit on amdgcn)
  unsigned lds_addr =
      (unsigned)(__UINTPTR_TYPE__)(__attribute__((address_space(3))) float*)l;
  unsigned long long gaddr = (unsigned long long)(__UINTPTR_TYPE__)g;
  // GV mode: vdst = LDS addr VGPR, vaddr = 64-bit global addr, no saddr.
  // 16 bytes per lane, tracked with ASYNCcnt.
  asm volatile("global_load_async_to_lds_b128 %0, %1, off"
               :
               : "v"(lds_addr), "v"(gaddr)
               : "memory");
}
#define ED_WAIT_ASYNC(n)                          \
  do {                                            \
    __builtin_amdgcn_s_wait_asynccnt(n);          \
    asm volatile("" ::: "memory");                \
  } while (0)
#endif

// Per-segment: sum-of-squares across the wave's 256 values, clamp-normalize,
// accumulate this lane's 8 dims.
__device__ __forceinline__ void ed_accum(const float4& u0, const float4& u1,
                                         float* acc) {
  float ss = u0.x * u0.x + u0.y * u0.y + u0.z * u0.z + u0.w * u0.w +
             u1.x * u1.x + u1.y * u1.y + u1.z * u1.z + u1.w * u1.w;
#pragma unroll
  for (int m = 16; m >= 1; m >>= 1) ss += __shfl_xor(ss, m, 32);  // wave32
  float inv = 1.0f / fmaxf(sqrtf(ss), ED_EPS);
  acc[0] += u0.x * inv; acc[1] += u0.y * inv;
  acc[2] += u0.z * inv; acc[3] += u0.w * inv;
  acc[4] += u1.x * inv; acc[5] += u1.y * inv;
  acc[6] += u1.z * inv; acc[7] += u1.w * inv;
}

// Stage 1: grid (P, 16). Each workgroup = 8 waves; each wave owns segments
// s = wg*8+wave, += P*8. Lane l covers dims [8l, 8l+8). Writes one 256-dim
// partial sum per tensor per WG into ws (no atomics -> deterministic).
__global__ __launch_bounds__(256) void ed_partial_kernel(
    const float* __restrict__ x1, const float* __restrict__ x2,
    float* __restrict__ ws1, float* __restrict__ ws2, int P) {
  const int tid = threadIdx.x;
  const int wave = tid >> 5;
  const int lane = tid & 31;
  const int wg = blockIdx.x;
  const int batch = blockIdx.y;
  const int wavesPerBatch = P << 3;
  const int gwave = (wg << 3) + wave;
  const size_t bbase = (size_t)batch * ED_S * ED_D;

  float acc1[8], acc2[8];
#pragma unroll
  for (int k = 0; k < 8; ++k) { acc1[k] = 0.0f; acc2[k] = 0.0f; }

  __shared__ float red[8][2 * ED_D];

#if ED_USE_ASYNC
  // Per-wave double-buffered staging: [wave][buf][tensor][256 floats] = 32 KB.
  __shared__ float stage[8][2][2][ED_D];
  int buf = 0;
  int s = gwave;
  if (s < ED_S) {
    const float* g1 = x1 + bbase + (size_t)s * ED_D + lane * 8;
    const float* g2 = x2 + bbase + (size_t)s * ED_D + lane * 8;
    ed_async16(g1,     &stage[wave][0][0][lane * 8]);
    ed_async16(g1 + 4, &stage[wave][0][0][lane * 8 + 4]);
    ed_async16(g2,     &stage[wave][0][1][lane * 8]);
    ed_async16(g2 + 4, &stage[wave][0][1][lane * 8 + 4]);
  }
  for (; s < ED_S; s += wavesPerBatch) {
    const int sn = s + wavesPerBatch;
    if (sn < ED_S) {  // prefetch next segment into the other buffer
      const float* g1 = x1 + bbase + (size_t)sn * ED_D + lane * 8;
      const float* g2 = x2 + bbase + (size_t)sn * ED_D + lane * 8;
      ed_async16(g1,     &stage[wave][buf ^ 1][0][lane * 8]);
      ed_async16(g1 + 4, &stage[wave][buf ^ 1][0][lane * 8 + 4]);
      ed_async16(g2,     &stage[wave][buf ^ 1][1][lane * 8]);
      ed_async16(g2 + 4, &stage[wave][buf ^ 1][1][lane * 8 + 4]);
      ED_WAIT_ASYNC(4);  // the 4 older ops (current buffer) are complete
    } else {
      ED_WAIT_ASYNC(0);
    }
    const float4 u0 = *(const float4*)&stage[wave][buf][0][lane * 8];
    const float4 u1 = *(const float4*)&stage[wave][buf][0][lane * 8 + 4];
    const float4 w0 = *(const float4*)&stage[wave][buf][1][lane * 8];
    const float4 w1 = *(const float4*)&stage[wave][buf][1][lane * 8 + 4];
    ed_accum(u0, u1, acc1);
    ed_accum(w0, w1, acc2);
    buf ^= 1;
  }
#else
  for (int s = gwave; s < ED_S; s += wavesPerBatch) {
    const float* g1 = x1 + bbase + (size_t)s * ED_D + lane * 8;
    const float* g2 = x2 + bbase + (size_t)s * ED_D + lane * 8;
    const float4 u0 = *(const float4*)g1;
    const float4 u1 = *(const float4*)(g1 + 4);
    const float4 w0 = *(const float4*)g2;
    const float4 w1 = *(const float4*)(g2 + 4);
    ed_accum(u0, u1, acc1);
    ed_accum(w0, w1, acc2);
  }
#endif

  // Cross-wave reduce inside the workgroup via LDS.
#pragma unroll
  for (int k = 0; k < 8; ++k) {
    red[wave][lane * 8 + k] = acc1[k];
    red[wave][ED_D + lane * 8 + k] = acc2[k];
  }
  __syncthreads();
  for (int j = tid; j < 2 * ED_D; j += 256) {
    float ssum = 0.0f;
#pragma unroll
    for (int w = 0; w < 8; ++w) ssum += red[w][j];
    float* dst = (j < ED_D) ? ws1 : ws2;
    dst[((size_t)batch * P + wg) * ED_D + (j & (ED_D - 1))] = ssum;
  }
}

// Stage 2: one workgroup. Reduce the P per-WG partials to the 16x256 sum
// vectors in LDS, then compute the 16x16 Gram matrix diag with 8 chained
// v_wmma_f32_16x16x32_f16 (K=256 total). All 8 waves run the WMMA redundantly
// so EXEC is all-1s; duplicate stores write identical values.
__global__ __launch_bounds__(256) void ed_final_kernel(
    const float* __restrict__ ws1, const float* __restrict__ ws2,
    float* __restrict__ out, int P) {
  __shared__ float mbuf[2][ED_B][ED_D];  // 32 KB
  const int tid = threadIdx.x;

  for (int j = tid; j < 2 * ED_B * ED_D; j += 256) {
    const int t = j >> 12;           // tensor
    const int rem = j & 4095;
    const int b = rem >> 8;          // batch
    const int d = rem & (ED_D - 1);  // dim
    const float* src = (t == 0) ? ws1 : ws2;
    float ssum = 0.0f;
    for (int p = 0; p < P; ++p) ssum += src[((size_t)b * P + p) * ED_D + d];
    mbuf[t][b][d] = ssum;
  }
  __syncthreads();

  const int lane = tid & 31;
  const int hi = lane >> 4;    // lane half
  const int m = lane & 15;     // = M row for A, N col for B
  v8f c = {};
#pragma unroll
  for (int chunk = 0; chunk < 8; ++chunk) {
    const int dbase = chunk * 32;
    v16h a, b;
#pragma unroll
    for (int e = 0; e < 16; ++e) {
      // ISA 7.12.2: 16-bit A 16x32 — lane<16 holds K {0..7,16..23},
      // lane>=16 holds K {8..15,24..31}; B 32x16 uses the same K order.
      const int ka = (e & 7) + ((e >> 3) << 4) + (hi << 3);
      const int kb = (e & 7) + ((e >> 3) << 4) + (hi << 3);
      a[e] = (_Float16)mbuf[0][m][dbase + ka];  // A[M=m][K=ka]
      b[e] = (_Float16)mbuf[1][m][dbase + kb];  // B[K=kb][N=m]
    }
    c = __builtin_amdgcn_wmma_f32_16x16x32_f16(
        /*neg_a=*/false, a, /*neg_b=*/false, b,
        /*c_mod=*/(short)0, c, /*reuse_a=*/false, /*reuse_b=*/false);
  }

  // D layout: VGPR r -> (M=r, N=lane) for lanes 0-15, (M=r+8, N=lane-16) for
  // lanes 16-31. Diagonal lives in lanes 0..7 (batch=lane) and 24..31
  // (batch=lane-16). Sums carry a 4096 factor each side.
  const float inv = 1.0f / (4096.0f * 4096.0f);
#pragma unroll
  for (int r = 0; r < 8; ++r) {
    const float v = 1.0f - c[r] * inv;
    if (lane == r)      out[r] = v;
    if (lane == r + 24) out[r + 8] = v;
  }
}

extern "C" void kernel_launch(void* const* d_in, const int* in_sizes, int n_in,
                              void* d_out, int out_size, void* d_ws,
                              size_t ws_size, hipStream_t stream) {
  (void)in_sizes; (void)n_in; (void)out_size;
  const float* x1 = (const float*)d_in[0];
  const float* x2 = (const float*)d_in[1];
  float* out = (float*)d_out;

  // ws holds 2 * 16 * P * 256 floats = P * 32768 bytes of per-WG partials.
  int P = (int)(ws_size / (size_t)(2 * ED_B * ED_D * sizeof(float)));
  if (P > 32) P = 32;
  if (P < 1) P = 1;
  float* ws1 = (float*)d_ws;
  float* ws2 = ws1 + (size_t)ED_B * P * ED_D;

  hipLaunchKernelGGL(ed_partial_kernel, dim3(P, ED_B), dim3(256), 0, stream,
                     x1, x2, ws1, ws2, P);
  hipLaunchKernelGGL(ed_final_kernel, dim3(1), dim3(256), 0, stream,
                     ws1, ws2, out, P);
}
